// att_block_13511967113769
// MI455X (gfx1250) — matrix operand order repaired and checked by reading, not compiled, not run on hardware
//
#include <hip/hip_runtime.h>
#include <hip/hip_bf16.h>
#include <math.h>

typedef __attribute__((ext_vector_type(16))) _Float16 v16h;
typedef __attribute__((ext_vector_type(8)))  float    v8f;
typedef unsigned int u32x4 __attribute__((ext_vector_type(4)));
typedef int          i32x8 __attribute__((ext_vector_type(8)));
typedef int          i32x4 __attribute__((ext_vector_type(4)));

#define LN_EPS 1e-5f

__device__ __forceinline__ float leaky01(float x) { return x > 0.f ? x : 0.01f * x; }

// ---------------------------------------------------------------------------
// Tensor Data Mover: 2-D tile load Global -> LDS (CDNA5 ISA ch.7 TDM, D# ch.8)
// data_size code: 0=1B 1=2B 2=4B 3=8B; dims/strides in elements.
// amdgpu-toolchain (clang-23) 6-arg builtin form. Completion via TENSORcnt.
// ---------------------------------------------------------------------------
__device__ __forceinline__ void tdm_load_2d(unsigned lds_addr, const void* gptr,
                                            unsigned dsz, unsigned tensor_d0,
                                            unsigned tensor_d1, unsigned tile_d0,
                                            unsigned tile_d1, unsigned stride0) {
  const unsigned long long ga = (unsigned long long)gptr;
  u32x4 g0;
  g0[0] = 1u;                                   // count=1 (valid user descriptor)
  g0[1] = lds_addr;                             // LDS byte address
  g0[2] = (unsigned)ga;                         // global_addr[31:0]
  g0[3] = (unsigned)(ga >> 32) | (2u << 30);    // global_addr[56:32] | type=2
  i32x8 g1;
  g1[0] = (int)(dsz << 16);                     // wg_mask=0, data_size
  g1[1] = (int)((tensor_d0 & 0xFFFFu) << 16);   // tensor_dim0[15:0] @bits63:48
  g1[2] = (int)((tensor_d0 >> 16) | ((tensor_d1 & 0xFFFFu) << 16));
  g1[3] = (int)((tensor_d1 >> 16) | (tile_d0 << 16));
  g1[4] = (int)(tile_d1 & 0xFFFFu);             // tile_dim1, tile_dim2=0
  g1[5] = (int)stride0;                         // tensor_dim0_stride[31:0]
  g1[6] = 0;
  g1[7] = 0;
  const i32x4 z4 = {0, 0, 0, 0};                // groups 2/3 unused (2-D tensor)
  const i32x8 z8 = {0, 0, 0, 0, 0, 0, 0, 0};
  __builtin_amdgcn_tensor_load_to_lds(g0, g1, z4, z4, z8, 0);
}

// ---- WMMA fragment builders (contiguous k order k = 16*(lane/16) + j, wave32) ----
// A 16x32: lane<16 row=lane K=0..15; lane>=16 row=lane-16 K=16..31
__device__ __forceinline__ v16h a_frag_h(const _Float16* base, int ld, int lane) {
  const int r = lane & 15;
  const int kb = (lane >> 4) * 16;
  const _Float16* p = base + r * ld + kb;
  v16h a;
#pragma unroll
  for (int j = 0; j < 16; ++j) a[j] = p[j];
  return a;
}

// B 32x16 from a K-contiguous [N][K] tile: b[j] = T[n][kb+j] (contiguous!)
__device__ __forceinline__ v16h b_frag_T(const _Float16* row, int lane) {
  // caller passes row = T + n*ldk + (lane>>4)*16 already
  v16h b;
#pragma unroll
  for (int j = 0; j < 16; ++j) b[j] = row[j];
  return b;
}

// ---------------------------------------------------------------------------
// Weight convert+transpose: WT16[n*512+k] = (f16) W[k*512+n]   (one-time)
// grid (16,16), 256 thr; 32x32 tiles through padded LDS
// ---------------------------------------------------------------------------
__global__ __launch_bounds__(256) void wconv_kernel(const float* __restrict__ W,
                                                    _Float16* __restrict__ WT) {
  __shared__ float t[32][33];
  const int tx = threadIdx.x & 31, ty = threadIdx.x >> 5;
  const int bn = blockIdx.x * 32, bk = blockIdx.y * 32;
#pragma unroll
  for (int i = 0; i < 4; ++i)
    t[ty + i * 8][tx] = W[(size_t)(bk + ty + i * 8) * 512 + bn + tx];
  __syncthreads();
#pragma unroll
  for (int i = 0; i < 4; ++i)
    WT[(size_t)(bn + ty + i * 8) * 512 + bk + tx] = (_Float16)t[tx][ty + i * 8];
}

// ---------------------------------------------------------------------------
// base embedding: out[n, m*128+e] = leaky((x@W1+b1)@W2+b2), m=0..3 (tile)
// writes f32 (residual path) and f16 (GEMM operand path)
// ---------------------------------------------------------------------------
template <int K1>
__global__ __launch_bounds__(128) void base_emb_kernel(
    const float* __restrict__ x, const float* __restrict__ W1,
    const float* __restrict__ b1, const float* __restrict__ W2,
    const float* __restrict__ b2, float* __restrict__ out,
    _Float16* __restrict__ out16, int N) {
  __shared__ float xs[16 * K1];
  __shared__ float hs[16 * 128];
  const int tid = threadIdx.x;
  const int r0 = blockIdx.x * 16;
  for (int i = tid; i < 16 * K1; i += 128) {
    int r = i / K1, k = i - r * K1;
    xs[i] = x[(size_t)(r0 + r) * K1 + k];
  }
  __syncthreads();
  {
    const int c0 = tid;
    const float bb = b1[c0];
    for (int r = 0; r < 16; ++r) {
      float s = bb;
      for (int k = 0; k < K1; ++k) s += xs[r * K1 + k] * W1[k * 128 + c0];
      hs[r * 128 + c0] = s;  // no activation between the two linears
    }
  }
  __syncthreads();
  {
    const int c0 = tid;
    const float bb = b2[c0];
    for (int r = 0; r < 16; ++r) {
      float s = bb;
      for (int k = 0; k < 128; ++k) s += hs[r * 128 + k] * W2[k * 128 + c0];
      s = leaky01(s);
      const size_t ro = (size_t)(r0 + r) * 512;
      const _Float16 sh = (_Float16)s;
#pragma unroll
      for (int m = 0; m < 4; ++m) {
        out[ro + m * 128 + c0] = s;
        out16[ro + m * 128 + c0] = sh;
      }
    }
  }
}

// ---------------------------------------------------------------------------
// Y[N,512] = leaky(X16[N,512] @ W + bias), W given transposed f16 (WT[n][k]).
// Pure-f16 operands; both tiles staged by the TDM. Optional f32/f16 outputs.
// block 256 (8 waves); tile M=128, N=64, K-step 32.
// ---------------------------------------------------------------------------
__global__ __launch_bounds__(256) void gemm512h_kernel(
    const _Float16* __restrict__ X, const _Float16* __restrict__ WT,
    const float* __restrict__ bias, float* __restrict__ Yf,
    _Float16* __restrict__ Yh, int N) {
  __shared__ _Float16 Xs[128 * 32];  // 8KB
  __shared__ _Float16 Ws[64 * 32];   // 4KB (output-col-major, K contiguous)
  const int tid = threadIdx.x;
  const int lane = tid & 31;
  const int wv = tid >> 5;
  const int m0 = blockIdx.y * 128;
  const int n0 = blockIdx.x * 64;
  v8f acc[4] = {};
  for (int k0 = 0; k0 < 512; k0 += 32) {
    if (wv == 0) {
      tdm_load_2d((unsigned)(uintptr_t)(void*)Xs, X + (size_t)m0 * 512 + k0,
                  1u, 512u, (unsigned)N, 32u, 128u, 512u);
      tdm_load_2d((unsigned)(uintptr_t)(void*)Ws, WT + (size_t)n0 * 512 + k0,
                  1u, 512u, 512u, 32u, 64u, 512u);
      __builtin_amdgcn_s_wait_tensorcnt(0);
    }
    __syncthreads();
    const v16h a = a_frag_h(Xs + wv * 16 * 32, 32, lane);
    const int hi = lane >> 4, nl = lane & 15;
#pragma unroll
    for (int nt = 0; nt < 4; ++nt) {
      const v16h b = b_frag_T(Ws + (nt * 16 + nl) * 32 + hi * 16, lane);
      acc[nt] = __builtin_amdgcn_wmma_f32_16x16x32_f16(false, a, false, b,
                                                       (short)0, acc[nt], false, false);
    }
    __syncthreads();
  }
  const int hi = lane >> 4, nl = lane & 15;
#pragma unroll
  for (int nt = 0; nt < 4; ++nt) {
    const int col = n0 + nt * 16 + nl;
    const float bv = bias[col];
#pragma unroll
    for (int r = 0; r < 8; ++r) {
      const int row = m0 + wv * 16 + hi * 8 + r;
      const float vv = leaky01(acc[nt][r] + bv);
      const size_t idx = (size_t)row * 512 + col;
      if (Yf) Yf[idx] = vv;
      if (Yh) Yh[idx] = (_Float16)vv;
    }
  }
}

// ---------------------------------------------------------------------------
// per-column mean/var (biased) over N rows of Y[N,512]
// ---------------------------------------------------------------------------
__global__ __launch_bounds__(256) void colstats_kernel(
    const float* __restrict__ Y, int N, float* __restrict__ mean,
    float* __restrict__ var) {
  __shared__ float ssum[256], ssq[256];
  const int tid = threadIdx.x;
  const int c = blockIdx.x * 32 + (tid & 31);
  const int rl = tid >> 5;
  float s = 0.f, q = 0.f;
  for (int r = rl; r < N; r += 8) {
    const float v = Y[(size_t)r * 512 + c];
    s += v; q += v * v;
  }
  ssum[tid] = s; ssq[tid] = q;
  __syncthreads();
  if (tid < 32) {
    float ts = 0.f, tq = 0.f;
#pragma unroll
    for (int i = 0; i < 8; ++i) { ts += ssum[i * 32 + tid]; tq += ssq[i * 32 + tid]; }
    const float m = ts / (float)N;
    mean[c] = m;
    var[c] = tq / (float)N - m * m;
  }
}

// BN epilogue emits f16: qe/ke/ve are consumed only by the WMMA attention.
__global__ void bn_apply_kernel(const float* __restrict__ Y,
                                const float* __restrict__ mean,
                                const float* __restrict__ var,
                                const float* __restrict__ g,
                                const float* __restrict__ b,
                                _Float16* __restrict__ E, int total) {
  const int i = blockIdx.x * blockDim.x + threadIdx.x;
  if (i >= total) return;
  const int c = i & 511;
  E[i] = (_Float16)(g[c] * (Y[i] - mean[c]) * rsqrtf(var[c] + LN_EPS) + b[c]);
}

// ---------------------------------------------------------------------------
// LayerNorm over last dim (512): out = ln(X + optional R); optional f16 copy
// ---------------------------------------------------------------------------
__global__ __launch_bounds__(256) void ln_kernel(
    const float* __restrict__ X, const float* __restrict__ R,
    const float* __restrict__ g, const float* __restrict__ b,
    float* __restrict__ out, _Float16* __restrict__ out16, int N) {
  __shared__ float sred[256];
  const int row = blockIdx.x;
  const int tid = threadIdx.x;
  const size_t ro = (size_t)row * 512;
  float v0 = X[ro + tid];
  float v1 = X[ro + tid + 256];
  if (R) { v0 += R[ro + tid]; v1 += R[ro + tid + 256]; }
  sred[tid] = v0 + v1;
  __syncthreads();
  for (int st = 128; st > 0; st >>= 1) {
    if (tid < st) sred[tid] += sred[tid + st];
    __syncthreads();
  }
  const float m = sred[0] * (1.f / 512.f);
  __syncthreads();
  const float d0 = v0 - m, d1 = v1 - m;
  sred[tid] = d0 * d0 + d1 * d1;
  __syncthreads();
  for (int st = 128; st > 0; st >>= 1) {
    if (tid < st) sred[tid] += sred[tid + st];
    __syncthreads();
  }
  const float inv = rsqrtf(sred[0] * (1.f / 512.f) + LN_EPS);
  const float o0 = g[tid] * d0 * inv + b[tid];
  const float o1 = g[tid + 256] * d1 * inv + b[tid + 256];
  out[ro + tid] = o0;
  out[ro + tid + 256] = o1;
  if (out16) { out16[ro + tid] = (_Float16)o0; out16[ro + tid + 256] = (_Float16)o1; }
}

// ---------------------------------------------------------------------------
// Fused masked attention (flash-style), f16 operands:
//   out = softmax_row(A o (qe @ ke^T) / sqrt(512)) @ ve + res
// block: 256 thr (8 waves), one 16-row q tile; key tiles of 128.
// ke tile streamed into LDS by the Tensor Data Mover in 128x128 f16 chunks.
// ---------------------------------------------------------------------------
__global__ __launch_bounds__(256) void attn_kernel(
    const _Float16* __restrict__ qe, const _Float16* __restrict__ ke,
    const _Float16* __restrict__ ve, const float* __restrict__ Amask,
    const float* __restrict__ res, float* __restrict__ out, int Nq, int Nk) {
  __shared__ _Float16 qsh[16 * 512];   // 16KB q tile
  __shared__ _Float16 ksh[128 * 128];  // 32KB ke chunk (keys x 128 feats)
  __shared__ float sS[16 * 128];       // 8KB masked scores
  __shared__ _Float16 pS[16 * 128];    // 4KB P tile (WMMA A operand)
  __shared__ float rowM[16], rowL[16], alpha[16], rowSum[16];

  const int tid = threadIdx.x;
  const int lane = tid & 31;
  const int wv = tid >> 5;
  const int hi = lane >> 4, nl = lane & 15;
  const int q0 = blockIdx.x * 16;
  const float scale = 0.04419417382415922f;  // 1/sqrt(512)

  for (int i = tid; i < 16 * 512 / 8; i += 256)
    ((float4*)qsh)[i] = ((const float4*)(qe + (size_t)q0 * 512))[i];
  if (tid < 16) { rowM[tid] = -1e30f; rowL[tid] = 0.f; }
  __syncthreads();

  v8f acc[4] = {};
  const int colBase = wv * 64;

  for (int kt = 0; kt < Nk; kt += 128) {
    if (kt + 128 < Nk)  // prefetch next mask tile (global_prefetch_b8)
      __builtin_prefetch(Amask + (size_t)(q0 + (tid >> 4)) * Nk + kt + 128 + ((tid & 15) << 3), 0, 1);

    // ---- S = qe @ ke^T for this wave's 16 keys; ke chunks via TDM ----
    v8f sfr = {};
#pragma unroll 1
    for (int fc = 0; fc < 4; ++fc) {
      __syncthreads();  // everyone done with previous ksh contents
      if (wv == 0) {
        tdm_load_2d((unsigned)(uintptr_t)(void*)ksh,
                    ke + (size_t)kt * 512 + fc * 128,
                    1u, 512u, (unsigned)Nk, 128u, 128u, 512u);
        __builtin_amdgcn_s_wait_tensorcnt(0);
      }
      __syncthreads();
#pragma unroll
      for (int kc = 0; kc < 4; ++kc) {
        const v16h a = a_frag_h(qsh + fc * 128 + kc * 32, 512, lane);
        const v16h b = b_frag_T(ksh + (wv * 16 + nl) * 128 + kc * 32 + hi * 16, lane);
        sfr = __builtin_amdgcn_wmma_f32_16x16x32_f16(false, a, false, b,
                                                     (short)0, sfr, false, false);
      }
    }
    // ---- mask-by-multiply, stash to LDS ----
    const int keyBase = kt + wv * 16;
#pragma unroll
    for (int r = 0; r < 8; ++r) {
      const int qrow = q0 + hi * 8 + r;
      const int key = keyBase + nl;
      const float mval = Amask[(size_t)qrow * Nk + key];
      sS[(hi * 8 + r) * 128 + wv * 16 + nl] = mval * (sfr[r] * scale);
    }
    __syncthreads();
    // ---- online softmax bookkeeping ----
    if (tid < 16) {
      float mx = -1e30f;
      const float* sr = sS + tid * 128;
      for (int k2 = 0; k2 < 128; ++k2) mx = fmaxf(mx, sr[k2]);
      const float newM = fmaxf(rowM[tid], mx);
      const float al = __expf(rowM[tid] - newM);
      alpha[tid] = al;
      rowM[tid] = newM;
      rowL[tid] *= al;
      rowSum[tid] = 0.f;
    }
    __syncthreads();
    {  // P = exp(S - newM) -> f16 tile; partial row sums
      const int r = tid >> 4;
      const int c0 = (tid & 15) * 8;
      const float rm = rowM[r];
      float part = 0.f;
#pragma unroll
      for (int j = 0; j < 8; ++j) {
        const float p = __expf(sS[r * 128 + c0 + j] - rm);
        pS[r * 128 + c0 + j] = (_Float16)p;
        part += p;
      }
      atomicAdd(&rowSum[r], part);
    }
#pragma unroll
    for (int nt = 0; nt < 4; ++nt)  // rescale running accumulators
#pragma unroll
      for (int r = 0; r < 8; ++r) acc[nt][r] *= alpha[hi * 8 + r];
    __syncthreads();
    if (tid < 16) rowL[tid] += rowSum[tid];
    // ---- O += P @ ve (f16 B^T frags straight from global; L2-resident) ----
#pragma unroll 1
    for (int kk = 0; kk < 4; ++kk) {
      const v16h a = a_frag_h(pS + kk * 32, 128, lane);
#pragma unroll
      for (int nt = 0; nt < 4; ++nt) {
        const _Float16* vp = ve + (size_t)(kt + kk * 32 + hi * 16) * 512 + colBase + nt * 16 + nl;
        v16h b;
#pragma unroll
        for (int j = 0; j < 16; ++j) b[j] = vp[(size_t)j * 512];
        acc[nt] = __builtin_amdgcn_wmma_f32_16x16x32_f16(false, a, false, b,
                                                         (short)0, acc[nt], false, false);
      }
    }
    __syncthreads();
  }
  // ---- normalize + residual ----
#pragma unroll
  for (int nt = 0; nt < 4; ++nt) {
    const int col = colBase + nt * 16 + nl;
#pragma unroll
    for (int r = 0; r < 8; ++r) {
      const int rloc = hi * 8 + r;
      const size_t ro = (size_t)(q0 + rloc) * 512;
      out[ro + col] = acc[nt][r] / rowL[rloc] + res[ro + col];
    }
  }
}

// ---------------------------------------------------------------------------
extern "C" void kernel_launch(void* const* d_in, const int* in_sizes, int n_in,
                              void* d_out, int out_size, void* d_ws, size_t ws_size,
                              hipStream_t stream) {
  const int NC = 4096, NV = 8192;
  const float* A  = (const float*)d_in[0];
  const float* AT = (const float*)d_in[1];
  const float* v  = (const float*)d_in[2];
  const float* c  = (const float*)d_in[3];
  const float* embv_W1 = (const float*)d_in[4];
  const float* embv_b1 = (const float*)d_in[5];
  const float* embv_W2 = (const float*)d_in[6];
  const float* embv_b2 = (const float*)d_in[7];
  const float* embc_W1 = (const float*)d_in[8];
  const float* embc_b1 = (const float*)d_in[9];
  const float* embc_W2 = (const float*)d_in[10];
  const float* embc_b2 = (const float*)d_in[11];
  const float* const* P = (const float* const*)d_in;
  const float* for1b = (const float*)d_in[49];
  const float* ln11g = (const float*)d_in[50]; const float* ln11b = (const float*)d_in[51];
  const float* ln12g = (const float*)d_in[52]; const float* ln12b = (const float*)d_in[53];
  const float* ln21g = (const float*)d_in[54]; const float* ln21b = (const float*)d_in[55];
  const float* ln22g = (const float*)d_in[56]; const float* ln22b = (const float*)d_in[57];

  // ---- workspace carve-up ----
  char* wsb = (char*)d_ws;
  size_t off = 0;
  auto carveF = [&](size_t n) { float* p = (float*)(wsb + off); off += n * 4; return p; };
  auto carveH = [&](size_t n) { _Float16* p = (_Float16*)(wsb + off); off += n * 2; return p; };
  float* c1t = carveF((size_t)NC * 512);
  float* v1t = carveF((size_t)NV * 512);
  float* h   = carveF((size_t)NV * 512);   // f32 ff outputs (LN residual input)
  float* y   = carveF((size_t)NV * 512);   // f32 pre-BN / LN f32 intermediates
  float* tmp = carveF((size_t)NV * 512);   // attention output (+residual)
  float* meanB = carveF(512);
  float* varB  = carveF(512);
  _Float16* c1th = carveH((size_t)NC * 512);
  _Float16* v1th = carveH((size_t)NV * 512);
  _Float16* h16  = carveH((size_t)NV * 512);  // f16 MLP hidden
  _Float16* lnh  = carveH((size_t)NV * 512);  // f16 LN outputs (GEMM operands)
  _Float16* qeB  = carveH((size_t)NV * 512);
  _Float16* keB  = carveH((size_t)NV * 512);
  _Float16* veB  = carveH((size_t)NV * 512);
  _Float16* wt16 = carveH((size_t)13 * 512 * 512);  // transposed f16 weights

  // ---- one-time weight convert+transpose (13 [512,512] matrices) ----
  const int wIdx[13] = {12, 14, 16, 18, 20, 22, 30, 32, 34, 36, 38, 40, 48};
  auto WTp = [&](int pidx) -> const _Float16* {
    for (int i = 0; i < 13; ++i)
      if (wIdx[i] == pidx) return wt16 + (size_t)i * 512 * 512;
    return wt16;
  };
  for (int i = 0; i < 13; ++i)
    wconv_kernel<<<dim3(16, 16), 256, 0, stream>>>(P[wIdx[i]], wt16 + (size_t)i * 512 * 512);

  auto gemmh = [&](const _Float16* X, int widx, const float* bb, float* Yf,
                   _Float16* Yh, int N) {
    dim3 g(8, N / 128);
    gemm512h_kernel<<<g, 256, 0, stream>>>(X, WTp(widx), bb, Yf, Yh, N);
  };
  auto mlpbn = [&](const _Float16* X16, int N, int base, int bnbase, _Float16* E) {
    gemmh(X16, base + 0, P[base + 1], nullptr, h16, N);   // layer1 -> f16
    gemmh(h16, base + 2, P[base + 3], y, nullptr, N);     // layer2 -> f32
    colstats_kernel<<<16, 256, 0, stream>>>(y, N, meanB, varB);
    const int total = N * 512;
    bn_apply_kernel<<<(total + 255) / 256, 256, 0, stream>>>(
        y, meanB, varB, P[bnbase + 0], P[bnbase + 1], E, total);
  };

  // ---- base embeddings (tiled x4 to D=512), f32 + f16 copies ----
  base_emb_kernel<32><<<NC / 16, 128, 0, stream>>>(c, embc_W1, embc_b1, embc_W2, embc_b2, c1t, c1th, NC);
  base_emb_kernel<64><<<NV / 16, 128, 0, stream>>>(v, embv_W1, embv_b1, embv_W2, embv_b2, v1t, v1th, NV);

  // ---- attention 1: q = c1t [NC], k = v1t [NV] ----
  mlpbn(c1th, NC, 12, 24, qeB);
  mlpbn(v1th, NV, 16, 26, keB);
  mlpbn(v1th, NV, 20, 28, veB);
  attn_kernel<<<NC / 16, 256, 0, stream>>>(qeB, keB, veB, A, c1t, tmp, NC, NV);
  float* c2i = y;  // f32 LN output (X input of ln12); y free here
  ln_kernel<<<NC, 256, 0, stream>>>(tmp, nullptr, ln11g, ln11b, c2i, lnh, NC);
  gemmh(lnh, 48, for1b, h, nullptr, NC);  // ff -> f32 h
  float* c2 = (float*)d_out;
  ln_kernel<<<NC, 256, 0, stream>>>(c2i, h, ln12g, ln12b, c2, lnh, NC);  // lnh := c2 f16

  // ---- attention 2: q = v1t [NV], k = c2 [NC] ----
  mlpbn(v1th, NV, 30, 42, qeB);
  mlpbn(lnh, NC, 34, 44, keB);
  mlpbn(lnh, NC, 38, 46, veB);
  attn_kernel<<<NV / 16, 256, 0, stream>>>(qeB, keB, veB, AT, v1t, tmp, NV, NC);
  float* v2i = y;  // y free here
  ln_kernel<<<NV, 256, 0, stream>>>(tmp, nullptr, ln21g, ln21b, v2i, lnh, NV);
  gemmh(lnh, 48, for1b, h, nullptr, NV);  // ff2 (reuses for1, faithful)
  float* v2 = (float*)d_out + (size_t)NC * 512;
  ln_kernel<<<NV, 256, 0, stream>>>(v2i, h, ln22g, ln22b, v2, nullptr, NV);

  (void)in_sizes; (void)n_in; (void)out_size; (void)ws_size;
}
